// cMultiHeadAttention_3040836846140
// MI455X (gfx1250) — matrix operand order repaired and checked by reading, not compiled, not run on hardware
//
#include <hip/hip_runtime.h>
#include <hip/hip_bf16.h>
#include <math.h>

// -------- CDNA5 WMMA types --------
typedef __attribute__((ext_vector_type(16))) __bf16 v16bf;
typedef __attribute__((ext_vector_type(8)))  __bf16 v8bf;
typedef __attribute__((ext_vector_type(8)))  float  v8f;
typedef __attribute__((ext_vector_type(4)))  int    v4i;

#define K_B  4
#define K_S  1024
#define K_E  1024
#define K_H  16
#define K_DH 64
#define M1   (K_B * K_S)   // 4096
#define N1   (3 * K_E)     // 3072
#define KK1  (K_E)         // 1024

static __device__ __forceinline__ v8f wmma_bf16(v16bf a, v16bf b, v8f c) {
  // D = A(16x32 bf16) * B(32x16 bf16) + C(16x16 f32)
  return __builtin_amdgcn_wmma_f32_16x16x32_bf16(false, a, false, b, (short)0, c,
                                                 false, false);
}

// ---- async global->LDS (CDNA5). Guarded: falls back to sync copy. ----
// Builtin signature (from probe diagnostic): param0 = v4i in AS1 (global),
// non-const; LDS side in AS3; then imm offset + imm cpol.
static __device__ __forceinline__ void async_b128(void* lds_dst, const void* gsrc) {
#if __has_builtin(__builtin_amdgcn_global_load_async_to_lds_b128)
  __builtin_amdgcn_global_load_async_to_lds_b128(
      (__attribute__((address_space(1))) v4i*)gsrc,
      (__attribute__((address_space(3))) v4i*)lds_dst, 0, 0);
#else
  *(uint4*)lds_dst = *(const uint4*)gsrc;
#endif
}

static __device__ __forceinline__ void wait_async() {
#if __has_builtin(__builtin_amdgcn_global_load_async_to_lds_b128)
#if __has_builtin(__builtin_amdgcn_s_wait_asynccnt)
  __builtin_amdgcn_s_wait_asynccnt(0);
#else
  asm volatile("s_wait_asynccnt 0" ::: "memory");
#endif
#endif
}

// A-operand (16x32, MxK) bf16 layout: lane L -> row m = L&15;
// VGPR0-3 hold K = kg+0..7, VGPR4-7 hold K = 16+kg+0..7, kg = (L<16)?0:8.
static __device__ __forceinline__ v16bf load_a_frag(const __bf16* base, int ld,
                                                    int lane, int k0) {
  const int m  = lane & 15;
  const int kg = (lane < 16) ? 0 : 8;
  const __bf16* p = base + (size_t)m * ld + k0 + kg;
  union { v16bf v; v8bf h[2]; } u;
  u.h[0] = *(const v8bf*)(p);
  u.h[1] = *(const v8bf*)(p + 16);
  return u.v;
}

// B-operand (32x16, KxN) bf16 layout: lane L -> col n = L&15;
// VGPR0-3 hold K = kg+0..7, VGPR4-7 hold K = 16+kg+0..7, kg = (L<16)?0:8. Source stored [N, K]
// row-major (weight rows / transposed V), so reads are contiguous 32B.
static __device__ __forceinline__ v16bf load_b_frag(const __bf16* base, int ld,
                                                    int lane, int k0) {
  const int n  = lane & 15;
  const int kb = (lane < 16) ? 0 : 8;
  const __bf16* p = base + (size_t)n * ld + k0 + kb;
  union { v16bf v; v8bf h[2]; } u;
  u.h[0] = *(const v8bf*)(p);
  u.h[1] = *(const v8bf*)(p + 16);
  return u.v;
}

// -------- fp32 -> bf16 conversion --------
__global__ void f32_to_bf16_kernel(const float* __restrict__ src,
                                   __bf16* __restrict__ dst, int n) {
  int i = blockIdx.x * blockDim.x + threadIdx.x;
  if (i < n) dst[i] = (__bf16)src[i];
}

// -------- complex GEMM: C = A * W^T + bias, A:[M,K], W:[N,K] (both r/i bf16)
// 4 waves/block share one 64-col W tile staged in LDS via async copies;
// wave w computes the 16-row M tile (bm*4+w).
// MODE 0: QKV epilogue -> q,k as [B,H,S,DH] bf16, v transposed [B,H,DH,S] bf16
// MODE 1: final epilogue -> interleaved complex64 (f32 re,im) into outc
template <int MODE>
__global__ void __launch_bounds__(128)
cgemm_bf16_wmma(const __bf16* __restrict__ Ar, const __bf16* __restrict__ Ai,
                const __bf16* __restrict__ Wr, const __bf16* __restrict__ Wi,
                const float* __restrict__ biasr, const float* __restrict__ biasi,
                int M, int N, int K,
                __bf16* __restrict__ qr, __bf16* __restrict__ qi,
                __bf16* __restrict__ kr, __bf16* __restrict__ ki,
                __bf16* __restrict__ vTr, __bf16* __restrict__ vTi,
                float* __restrict__ outc) {
  __shared__ __bf16 wstage[2][2][64][32];  // [buf][r/i][n][k]  16KB
  const int tid  = threadIdx.x;
  const int lane = tid & 31;
  const int wave = tid >> 5;
  const int mb   = M >> 6;                 // M/64 blocks along M
  const int bm   = blockIdx.x % mb;
  const int tn   = blockIdx.x / mb;        // 64-wide N tile, shared by block
  const int tm   = bm * 4 + wave;          // 16-row M tile per wave

  const size_t wbase = (size_t)(tn * 64) * K;
  const __bf16* Arb = Ar + (size_t)(tm * 16) * K;
  const __bf16* Aib = Ai + (size_t)(tm * 16) * K;

  // async-stage one 64x32 (r+i) W tile: 512 x 16B chunks, 4 per thread
  auto issue_w = [&](int buf, int k0) {
#pragma unroll
    for (int j = 0; j < 4; j++) {
      const int c   = tid + j * 128;       // 0..511
      const int row = c >> 2, ch = c & 3;  // 4 chunks per 32-k row
      const int ri  = row >> 6, n = row & 63;
      const __bf16* src = (ri ? Wi : Wr) + wbase + (size_t)n * K + k0 + ch * 8;
      async_b128(&wstage[buf][ri][n][ch * 8], src);
    }
  };

  const v8f z = {0.f, 0.f, 0.f, 0.f, 0.f, 0.f, 0.f, 0.f};
  v8f acc[4][4];  // [n-subtile][rr, ii, ri, ir]
#pragma unroll
  for (int t = 0; t < 4; t++)
#pragma unroll
    for (int g = 0; g < 4; g++) acc[t][g] = z;

  issue_w(0, 0);
  wait_async();
  __syncthreads();

  for (int k0 = 0; k0 < K; k0 += 32) {
    const int buf = (k0 >> 5) & 1;
    if (k0 + 32 < K) issue_w(buf ^ 1, k0 + 32);

    v16bf afr = load_a_frag(Arb, K, lane, k0);
    v16bf afi = load_a_frag(Aib, K, lane, k0);
#pragma unroll
    for (int t = 0; t < 4; t++) {
      v16bf wfr = load_b_frag(&wstage[buf][0][t * 16][0], 32, lane, 0);
      v16bf wfi = load_b_frag(&wstage[buf][1][t * 16][0], 32, lane, 0);
      acc[t][0] = wmma_bf16(afr, wfr, acc[t][0]);
      acc[t][1] = wmma_bf16(afi, wfi, acc[t][1]);
      acc[t][2] = wmma_bf16(afr, wfi, acc[t][2]);
      acc[t][3] = wmma_bf16(afi, wfr, acc[t][3]);
    }
    wait_async();
    __syncthreads();
  }

  // C layout: VGPR v, lanes 0-15 -> row v, lanes 16-31 -> row v+8; col = lane&15
  const int half8 = (lane < 16) ? 0 : 8;
  const int col_l = lane & 15;
#pragma unroll
  for (int t = 0; t < 4; t++) {
    const int n = tn * 64 + t * 16 + col_l;
    const float br = biasr[n];
    const float bi = biasi[n];
#pragma unroll
    for (int v = 0; v < 8; v++) {
      const int   m  = tm * 16 + half8 + v;
      const float cr = acc[t][0][v] - acc[t][1][v] + br;
      const float ci = acc[t][2][v] + acc[t][3][v] + bi;
      if (MODE == 0) {
        const int bidx = m / K_S, s = m % K_S;
        const int sec = n / K_E, e = n % K_E;
        const int h = e / K_DH, d = e % K_DH;
        const size_t bh = (size_t)bidx * K_H + h;
        if (sec == 0) {
          const size_t idx = (bh * K_S + s) * K_DH + d;
          qr[idx] = (__bf16)cr; qi[idx] = (__bf16)ci;
        } else if (sec == 1) {
          const size_t idx = (bh * K_S + s) * K_DH + d;
          kr[idx] = (__bf16)cr; ki[idx] = (__bf16)ci;
        } else {
          const size_t idx = (bh * K_DH + d) * K_S + s;  // V stored transposed
          vTr[idx] = (__bf16)cr; vTi[idx] = (__bf16)ci;
        }
      } else {
        const size_t idx = ((size_t)m * N + n) * 2;  // complex64 interleave
        outc[idx]     = cr;
        outc[idx + 1] = ci;
      }
    }
  }
}

// -------- flash-style complex attention --------
// 4 waves/block = 4 consecutive q-tiles of the SAME (b,h); K/V key blocks are
// async-staged into double-buffered LDS once per block and consumed by all
// 4 waves. Dual (re/im) online softmax; P->A-frag via per-wave LDS tile.
__global__ void __launch_bounds__(128)
cmha_flash_wmma(const __bf16* __restrict__ qr, const __bf16* __restrict__ qi,
                const __bf16* __restrict__ kr, const __bf16* __restrict__ ki,
                const __bf16* __restrict__ vTr, const __bf16* __restrict__ vTi,
                __bf16* __restrict__ aor, __bf16* __restrict__ aoi) {
  __shared__ __bf16 kstage[2][2][32][64];  // [buf][r/i][key][d]  16KB
  __shared__ __bf16 vstage[2][2][64][32];  // [buf][r/i][d][key]  16KB
  __shared__ __bf16 plds[4][2][16][32];    // [wave][r/i][row][key] 8KB
  const int tid  = threadIdx.x;
  const int lane = tid & 31;
  const int wave = tid >> 5;
  const int bh   = blockIdx.x / 16;            // 16 blocks per (b,h)
  const int qt   = (blockIdx.x % 16) * 4 + wave;
  const int b = bh / K_H, h = bh % K_H;
  const int s0 = qt * 16;

  const __bf16* qrb = qr  + ((size_t)bh * K_S + s0) * K_DH;
  const __bf16* qib = qi  + ((size_t)bh * K_S + s0) * K_DH;
  const __bf16* krb = kr  + (size_t)bh * K_S * K_DH;
  const __bf16* kib = ki  + (size_t)bh * K_S * K_DH;
  const __bf16* vrb = vTr + (size_t)bh * K_DH * K_S;
  const __bf16* vib = vTi + (size_t)bh * K_DH * K_S;

  // async-stage one 32-key block of K ([key][d]) and V^T ([d][key]):
  // K: 2*32 rows x 8 chunks = 512; V: 2*64 rows x 4 chunks = 512; 8/thread.
  auto issue_kv = [&](int buf, int kb) {
#pragma unroll
    for (int j = 0; j < 4; j++) {
      const int c = tid + j * 128;  // 0..511
      {
        const int row = c >> 3, ch = c & 7;
        const int ri = row >> 5, key = row & 31;
        const __bf16* src = (ri ? kib : krb) + (size_t)(kb + key) * K_DH + ch * 8;
        async_b128(&kstage[buf][ri][key][ch * 8], src);
      }
      {
        const int row = c >> 2, ch = c & 3;
        const int ri = row >> 6, d = row & 63;
        const __bf16* src = (ri ? vib : vrb) + (size_t)d * K_S + kb + ch * 8;
        async_b128(&vstage[buf][ri][d][ch * 8], src);
      }
    }
  };

  v16bf qfr[2], qfi[2];  // DH=64 -> 2 k-steps of 32 (kept in VGPRs all loop)
#pragma unroll
  for (int ks = 0; ks < 2; ks++) {
    qfr[ks] = load_a_frag(qrb, K_DH, lane, ks * 32);
    qfi[ks] = load_a_frag(qib, K_DH, lane, ks * 32);
  }

  const v8f z = {0.f, 0.f, 0.f, 0.f, 0.f, 0.f, 0.f, 0.f};
  v8f o[4][4];  // [{P_r*V_r, P_r*V_i, P_i*V_r, P_i*V_i}][dh-subtile]
#pragma unroll
  for (int g = 0; g < 4; g++)
#pragma unroll
    for (int j = 0; j < 4; j++) o[g][j] = z;

  float mr[8], mi[8], lr[8], li[8];
#pragma unroll
  for (int v = 0; v < 8; v++) { mr[v] = mi[v] = -1e30f; lr[v] = li[v] = 0.f; }

  const float scale = 0.125f;  // 1/sqrt(64)

  issue_kv(0, 0);
  wait_async();
  __syncthreads();

  for (int kb = 0; kb < K_S; kb += 32) {
    const int buf = (kb >> 5) & 1;
    if (kb + 32 < K_S) issue_kv(buf ^ 1, kb + 32);

    v8f sa[2][4];  // [key-subtile][rr, ii, ri, ir]
#pragma unroll
    for (int t = 0; t < 2; t++)
#pragma unroll
      for (int g = 0; g < 4; g++) sa[t][g] = z;

#pragma unroll
    for (int ks = 0; ks < 2; ks++) {
#pragma unroll
      for (int t = 0; t < 2; t++) {
        v16bf kfr = load_b_frag(&kstage[buf][0][t * 16][0], 64, lane, ks * 32);
        v16bf kfi = load_b_frag(&kstage[buf][1][t * 16][0], 64, lane, ks * 32);
        sa[t][0] = wmma_bf16(qfr[ks], kfr, sa[t][0]);
        sa[t][1] = wmma_bf16(qfi[ks], kfi, sa[t][1]);
        sa[t][2] = wmma_bf16(qfr[ks], kfi, sa[t][2]);
        sa[t][3] = wmma_bf16(qfi[ks], kfr, sa[t][3]);
      }
    }

    float sr[2][8], si[2][8];
#pragma unroll
    for (int t = 0; t < 2; t++)
#pragma unroll
      for (int v = 0; v < 8; v++) {
        sr[t][v] = (sa[t][0][v] - sa[t][1][v]) * scale;
        si[t][v] = (sa[t][2][v] + sa[t][3][v]) * scale;
      }

    // dual online softmax; row v lives on a 16-lane half, reduce with xor<=8
#pragma unroll
    for (int v = 0; v < 8; v++) {
      float bmr = fmaxf(sr[0][v], sr[1][v]);
      float bmi = fmaxf(si[0][v], si[1][v]);
#pragma unroll
      for (int off = 8; off >= 1; off >>= 1) {
        bmr = fmaxf(bmr, __shfl_xor(bmr, off, 32));
        bmi = fmaxf(bmi, __shfl_xor(bmi, off, 32));
      }
      const float nmr = fmaxf(mr[v], bmr);
      const float nmi = fmaxf(mi[v], bmi);
      const float alr = __expf(mr[v] - nmr);
      const float ali = __expf(mi[v] - nmi);
      mr[v] = nmr; mi[v] = nmi;
      float rsr = 0.f, rsi = 0.f;
#pragma unroll
      for (int t = 0; t < 2; t++) {
        const float pr = __expf(sr[t][v] - nmr);
        const float pi = __expf(si[t][v] - nmi);
        sr[t][v] = pr; si[t][v] = pi;
        rsr += pr; rsi += pi;
      }
#pragma unroll
      for (int off = 8; off >= 1; off >>= 1) {
        rsr += __shfl_xor(rsr, off, 32);
        rsi += __shfl_xor(rsi, off, 32);
      }
      lr[v] = lr[v] * alr + rsr;
      li[v] = li[v] * ali + rsi;
#pragma unroll
      for (int j = 0; j < 4; j++) {
        o[0][j][v] *= alr; o[1][j][v] *= alr;  // groups fed by P_r
        o[2][j][v] *= ali; o[3][j][v] *= ali;  // groups fed by P_i
      }
    }

    // P (C-layout f32) -> LDS -> A-fragment (bf16)
    const int rbase = (lane < 16) ? 0 : 8;
    const int cl = lane & 15;
#pragma unroll
    for (int t = 0; t < 2; t++)
#pragma unroll
      for (int v = 0; v < 8; v++) {
        plds[wave][0][rbase + v][t * 16 + cl] = (__bf16)sr[t][v];
        plds[wave][1][rbase + v][t * 16 + cl] = (__bf16)si[t][v];
      }
    __syncthreads();
    v16bf pfr = load_a_frag(&plds[wave][0][0][0], 32, lane, 0);
    v16bf pfi = load_a_frag(&plds[wave][1][0][0], 32, lane, 0);

#pragma unroll
    for (int j = 0; j < 4; j++) {
      v16bf vfr = load_b_frag(&vstage[buf][0][j * 16][0], 32, lane, 0);
      v16bf vfi = load_b_frag(&vstage[buf][1][j * 16][0], 32, lane, 0);
      o[0][j] = wmma_bf16(pfr, vfr, o[0][j]);
      o[1][j] = wmma_bf16(pfr, vfi, o[1][j]);
      o[2][j] = wmma_bf16(pfi, vfr, o[2][j]);
      o[3][j] = wmma_bf16(pfi, vfi, o[3][j]);
    }
    wait_async();
    __syncthreads();
  }

  // out_r = O_rr/l_r - O_ii/l_i ; out_i = O_ri/l_r + O_ir/l_i
#pragma unroll
  for (int v = 0; v < 8; v++) {
    const float invr = 1.f / lr[v];
    const float invi = 1.f / li[v];
    const int srow = s0 + ((lane < 16) ? 0 : 8) + v;
#pragma unroll
    for (int j = 0; j < 4; j++) {
      const int e = h * K_DH + j * 16 + (lane & 15);
      const float outr = o[0][j][v] * invr - o[3][j][v] * invi;
      const float outi = o[1][j][v] * invr + o[2][j][v] * invi;
      const size_t idx = ((size_t)b * K_S + srow) * K_E + e;
      aor[idx] = (__bf16)outr;
      aoi[idx] = (__bf16)outi;
    }
  }
}

extern "C" void kernel_launch(void* const* d_in, const int* in_sizes, int n_in,
                              void* d_out, int out_size, void* d_ws, size_t ws_size,
                              hipStream_t stream) {
  const float* x_r  = (const float*)d_in[0];
  const float* x_i  = (const float*)d_in[1];
  const float* wp_r = (const float*)d_in[2];
  const float* wp_i = (const float*)d_in[3];
  const float* bp_r = (const float*)d_in[4];
  const float* bp_i = (const float*)d_in[5];
  const float* wo_r = (const float*)d_in[6];
  const float* wo_i = (const float*)d_in[7];
  const float* bo_r = (const float*)d_in[8];
  const float* bo_i = (const float*)d_in[9];

  const size_t nX  = (size_t)K_B * K_S * K_E;         // 4Mi
  const size_t nWp = (size_t)3 * K_E * K_E;           // 3Mi
  const size_t nWo = (size_t)K_E * K_E;               // 1Mi
  const size_t nH  = (size_t)K_B * K_H * K_S * K_DH;  // 4Mi

  // workspace layout (~96MB of bf16 scratch)
  char* ws = (char*)d_ws;
  size_t off = 0;
  auto take = [&](size_t elems) {
    __bf16* p = (__bf16*)(ws + off);
    off += ((elems * sizeof(__bf16)) + 255) & ~(size_t)255;
    return p;
  };
  __bf16* Axr = take(nX);   __bf16* Axi = take(nX);
  __bf16* Wpr = take(nWp);  __bf16* Wpi = take(nWp);
  __bf16* Wor = take(nWo);  __bf16* Woi = take(nWo);
  __bf16* Qr  = take(nH);   __bf16* Qi  = take(nH);
  __bf16* Kr  = take(nH);   __bf16* Ki  = take(nH);
  __bf16* Vr  = take(nH);   __bf16* Vi  = take(nH);   // transposed [B,H,DH,S]
  __bf16* AOr = take(nX);   __bf16* AOi = take(nX);   // attention out [B,S,E]

  const int CT = 256;
  f32_to_bf16_kernel<<<(int)((nX  + CT - 1) / CT), CT, 0, stream>>>(x_r,  Axr, (int)nX);
  f32_to_bf16_kernel<<<(int)((nX  + CT - 1) / CT), CT, 0, stream>>>(x_i,  Axi, (int)nX);
  f32_to_bf16_kernel<<<(int)((nWp + CT - 1) / CT), CT, 0, stream>>>(wp_r, Wpr, (int)nWp);
  f32_to_bf16_kernel<<<(int)((nWp + CT - 1) / CT), CT, 0, stream>>>(wp_i, Wpi, (int)nWp);
  f32_to_bf16_kernel<<<(int)((nWo + CT - 1) / CT), CT, 0, stream>>>(wo_r, Wor, (int)nWo);
  f32_to_bf16_kernel<<<(int)((nWo + CT - 1) / CT), CT, 0, stream>>>(wo_i, Woi, (int)nWo);

  // QKV projection: [4096,1024] x [3072,1024]^T ; blocks = (M/64)*(N/64)
  cgemm_bf16_wmma<0><<<(M1 / 64) * (N1 / 64), 128, 0, stream>>>(
      Axr, Axi, Wpr, Wpi, bp_r, bp_i, M1, N1, KK1,
      Qr, Qi, Kr, Ki, Vr, Vi, nullptr);

  // flash complex attention: B*H*(S/16) = 4096 q-tiles, 4 waves/block
  cmha_flash_wmma<<<(K_B * K_H * (K_S / 16)) / 4, 128, 0, stream>>>(
      Qr, Qi, Kr, Ki, Vr, Vi, AOr, AOi);

  // output projection -> interleaved complex64 in d_out
  cgemm_bf16_wmma<1><<<(M1 / 64) * (K_E / 64), 128, 0, stream>>>(
      AOr, AOi, Wor, Woi, bo_r, bo_i, M1, K_E, KK1,
      nullptr, nullptr, nullptr, nullptr, nullptr, nullptr,
      (float*)d_out);
}
